// ClusterContrastiveLoss_68771016344107
// MI455X (gfx1250) — compile-verified, matrix operand order checked by reading, not run on hardware
//
#include <hip/hip_runtime.h>

#define NROWS 65536
#define KDIM  256
#define CHUNK 1024
#define LARGE_NUM 1e9f
#define EPS 1e-8f

typedef __attribute__((ext_vector_type(16))) __bf16 v16bf;
typedef __attribute__((ext_vector_type(8)))  __bf16 v8bf;
typedef __attribute__((ext_vector_type(8)))  float  v8f;

static __device__ __forceinline__ unsigned int f2bf(float f) {
    unsigned int u = __float_as_uint(f);
    return (u + 0x7FFFu + ((u >> 16) & 1u)) >> 16;   // round-to-nearest-even
}

static __device__ __forceinline__ float wredsum(float v) {
    #pragma unroll
    for (int m = 16; m > 0; m >>= 1) v += __shfl_xor(v, m, 32);
    return v;
}
static __device__ __forceinline__ float wredmax(float v) {
    #pragma unroll
    for (int m = 16; m > 0; m >>= 1) v = fmaxf(v, __shfl_xor(v, m, 32));
    return v;
}

static __device__ __forceinline__ v16bf load_frag(const unsigned short* p) {
    v8bf lo = *(const v8bf*)(p);
    v8bf hi = *(const v8bf*)(p + 16);
    v16bf r;
    #pragma unroll
    for (int i = 0; i < 8; ++i) { r[i] = lo[i]; r[i + 8] = hi[i]; }
    return r;
}

// ---------------------------------------------------------------------------
// Stage 0: zero the f32 accumulators in workspace
// ---------------------------------------------------------------------------
__global__ void ccl_zero_kernel(float* p, int n) {
    int i = blockIdx.x * blockDim.x + threadIdx.x;
    int stride = gridDim.x * blockDim.x;
    for (; i < n; i += stride) p[i] = 0.0f;
}

// ---------------------------------------------------------------------------
// Stage 1: row softmax of Q [N,256]; write transposed bf16 probs PT [256,N];
// accumulate column sum and column sum-of-squares (f32, via LDS + atomics).
// One wave per row; block = 8 waves = 32 rows.
// ---------------------------------------------------------------------------
__global__ void ccl_softmax_T_kernel(const float* __restrict__ Q,
                                     unsigned short* __restrict__ PT,
                                     float* __restrict__ colsum,
                                     float* __restrict__ colsumsq) {
    __shared__ float tile[32][264];      // 32 rows x 256 cols (+pad)
    __shared__ float s_cs[256];
    __shared__ float s_csq[256];

    const int t    = threadIdx.x;
    const int lane = t & 31;
    const int w    = t >> 5;
    const int n0   = blockIdx.x * 32;

    s_cs[t]  = 0.0f;
    s_csq[t] = 0.0f;
    __syncthreads();

    float cs[8], csq[8];
    #pragma unroll
    for (int i = 0; i < 8; ++i) { cs[i] = 0.0f; csq[i] = 0.0f; }

    #pragma unroll
    for (int rr = 0; rr < 4; ++rr) {
        const int r = w * 4 + rr;            // row within tile
        const int n = n0 + r;                // global row
        const float4* f = (const float4*)(Q + (size_t)n * KDIM + lane * 8);
        float4 x0 = f[0];
        float4 x1 = f[1];
        float x[8] = {x0.x, x0.y, x0.z, x0.w, x1.x, x1.y, x1.z, x1.w};

        float m = x[0];
        #pragma unroll
        for (int i = 1; i < 8; ++i) m = fmaxf(m, x[i]);
        m = wredmax(m);

        float e[8], s = 0.0f;
        #pragma unroll
        for (int i = 0; i < 8; ++i) { e[i] = __expf(x[i] - m); s += e[i]; }
        s = wredsum(s);
        const float inv = 1.0f / s;

        #pragma unroll
        for (int i = 0; i < 8; ++i) {
            float p = e[i] * inv;
            tile[r][lane * 8 + i] = p;
            cs[i]  += p;
            csq[i] += p * p;
        }
    }

    #pragma unroll
    for (int i = 0; i < 8; ++i) {
        atomicAdd(&s_cs[lane * 8 + i], cs[i]);      // ds_add_f32
        atomicAdd(&s_csq[lane * 8 + i], csq[i]);
    }
    __syncthreads();

    // transposed bf16 store: thread t owns cluster column k = t,
    // writes PT[k, n0..n0+31] as 4 x b128 stores (contiguous 64B)
    {
        const int k = t;
        unsigned int u[16];
        #pragma unroll
        for (int i = 0; i < 16; ++i) {
            unsigned int lo = f2bf(tile[2 * i + 0][k]);
            unsigned int hi = f2bf(tile[2 * i + 1][k]);
            u[i] = lo | (hi << 16);
        }
        uint4* dst = (uint4*)(PT + (size_t)k * NROWS + n0);
        #pragma unroll
        for (int i = 0; i < 4; ++i)
            dst[i] = make_uint4(u[4 * i], u[4 * i + 1], u[4 * i + 2], u[4 * i + 3]);

        atomicAdd(&colsum[k],  s_cs[k]);
        atomicAdd(&colsumsq[k], s_csq[k]);
    }
}

// ---------------------------------------------------------------------------
// Stage 2: partial Gram  G += X^T * Y  over one N-chunk using bf16 WMMA.
// XT, YT are transposed probs [256, N] bf16 (row-major, contiguous in n).
// Grid: (N/CHUNK, 8). Each block owns a 128x64 output block; each of the
// 8 waves computes a 2x2 group of 16x16 WMMA tiles, so every step loads
// 2 A-fragments + 2 B-fragments and issues 4 WMMAs (1.0 frag-loads/MMA).
// Partials accumulated into G with f32 atomics.
// ---------------------------------------------------------------------------
__global__ void ccl_gram_kernel(const unsigned short* __restrict__ XT,
                                const unsigned short* __restrict__ YT,
                                float* __restrict__ G) {
    const int t     = threadIdx.x;
    const int lane  = t & 31;
    const int wave  = t >> 5;
    const int m     = lane & 15;     // fragment row (A) / column (B,C)
    const int khalf = lane >> 4;     // which K-half this lane holds

    // 8x4 tile grid over a 128x64 block; wave owns 2x2 tiles
    const int tr0 = (wave & 3) * 2;          // first of 2 tile rows
    const int tc0 = (wave >> 2) * 2;         // first of 2 tile cols
    const int I00 = (blockIdx.y & 1) * 128 + (tr0 + 0) * 16;
    const int I01 = (blockIdx.y & 1) * 128 + (tr0 + 1) * 16;
    const int J00 = (blockIdx.y >> 1) * 64 + (tc0 + 0) * 16;
    const int J01 = (blockIdx.y >> 1) * 64 + (tc0 + 1) * 16;

    const size_t rowA0 = (size_t)(I00 + m) * NROWS;
    const size_t rowA1 = (size_t)(I01 + m) * NROWS;
    const size_t rowB0 = (size_t)(J00 + m) * NROWS;
    const size_t rowB1 = (size_t)(J01 + m) * NROWS;
    const int n0base = blockIdx.x * CHUNK;

    v8f acc00 = {};
    v8f acc01 = {};
    v8f acc10 = {};
    v8f acc11 = {};

    for (int n0 = n0base; n0 < n0base + CHUNK; n0 += 32) {
        const unsigned short* pa0 = XT + rowA0 + n0 + khalf * 8;
        const unsigned short* pa1 = XT + rowA1 + n0 + khalf * 8;
        const unsigned short* pb0 = YT + rowB0 + n0 + khalf * 8;
        const unsigned short* pb1 = YT + rowB1 + n0 + khalf * 8;

        // near-scope prefetch ~1KB ahead (data is L2-resident, reused 3x)
        __builtin_prefetch(pa0 + 512, 0, 3);
        __builtin_prefetch(pb0 + 512, 0, 3);

        v16bf a0 = load_frag(pa0);
        v16bf a1 = load_frag(pa1);
        v16bf b0 = load_frag(pb0);
        v16bf b1 = load_frag(pb1);

        acc00 = __builtin_amdgcn_wmma_f32_16x16x32_bf16(
                    false, a0, false, b0, (short)0, acc00, false, false);
        acc01 = __builtin_amdgcn_wmma_f32_16x16x32_bf16(
                    false, a0, false, b1, (short)0, acc01, false, false);
        acc10 = __builtin_amdgcn_wmma_f32_16x16x32_bf16(
                    false, a1, false, b0, (short)0, acc10, false, false);
        acc11 = __builtin_amdgcn_wmma_f32_16x16x32_bf16(
                    false, a1, false, b1, (short)0, acc11, false, false);
    }

    // C layout: VGPR r, lanes 0-15 -> M=r, lanes 16-31 -> M=r+8; col = lane%16
    #pragma unroll
    for (int r = 0; r < 8; ++r) {
        const int row0 = I00 + r + 8 * khalf;
        const int row1 = I01 + r + 8 * khalf;
        atomicAdd(&G[row0 * KDIM + J00 + m], acc00[r]);
        atomicAdd(&G[row0 * KDIM + J01 + m], acc01[r]);
        atomicAdd(&G[row1 * KDIM + J00 + m], acc10[r]);
        atomicAdd(&G[row1 * KDIM + J01 + m], acc11[r]);
    }
}

// ---------------------------------------------------------------------------
// Stage 3: epilogue — norms, entropy term, two 256x512 cross-entropies.
// Single block of 256 threads (8 waves); wave per row, lane-parallel columns.
// ---------------------------------------------------------------------------
__global__ void ccl_loss_kernel(const float* __restrict__ GAA,
                                const float* __restrict__ GAB,
                                const float* __restrict__ GBB,
                                const float* __restrict__ cs_a,
                                const float* __restrict__ csq_a,
                                const float* __restrict__ cs_b,
                                const float* __restrict__ csq_b,
                                float* __restrict__ out) {
    __shared__ float red[256];
    __shared__ float inva[256];
    __shared__ float invb[256];

    const int t    = threadIdx.x;
    const int lane = t & 31;
    const int w    = t >> 5;

    // ---- ne_loss: sum p log p for both distributions ----
    float sa = cs_a[t];
    red[t] = sa; __syncthreads();
    for (int s = 128; s > 0; s >>= 1) { if (t < s) red[t] += red[t + s]; __syncthreads(); }
    const float totA = red[0]; __syncthreads();
    float pa = sa / totA;
    red[t] = pa * __logf(pa); __syncthreads();
    for (int s = 128; s > 0; s >>= 1) { if (t < s) red[t] += red[t + s]; __syncthreads(); }
    const float neA = red[0]; __syncthreads();

    float sb = cs_b[t];
    red[t] = sb; __syncthreads();
    for (int s = 128; s > 0; s >>= 1) { if (t < s) red[t] += red[t + s]; __syncthreads(); }
    const float totB = red[0]; __syncthreads();
    float pb = sb / totB;
    red[t] = pb * __logf(pb); __syncthreads();
    for (int s = 128; s > 0; s >>= 1) { if (t < s) red[t] += red[t + s]; __syncthreads(); }
    const float neB = red[0]; __syncthreads();
    const float ne_loss = neA + neB;

    // ---- inverse clamped column norms ----
    inva[t] = 1.0f / fmaxf(__fsqrt_rn(csq_a[t]), EPS);
    invb[t] = 1.0f / fmaxf(__fsqrt_rn(csq_b[t]), EPS);
    __syncthreads();

    // ---- cross entropies over [AB | AA] and [BA | BB], labels = diag ----
    float accA = 0.0f, accB = 0.0f;
    for (int i = w; i < KDIM; i += 8) {
        const float ia = inva[i];
        const float ib = invb[i];
        float va[16], vb[16];
        float ma = -3.4e38f, mb = -3.4e38f;
        #pragma unroll
        for (int jj = 0; jj < 16; ++jj) {
            const int c = lane + 32 * jj;    // 0..511
            float xa, xb;
            if (c < KDIM) {
                xa = GAB[i * KDIM + c] * ia * invb[c];   // logits_ab
                xb = GAB[c * KDIM + i] * ib * inva[c];   // logits_ba
            } else {
                const int cc = c - KDIM;
                const float mask = (cc == i) ? LARGE_NUM : 0.0f;
                xa = GAA[i * KDIM + cc] * ia * inva[cc] - mask;   // logits_aa
                xb = GBB[i * KDIM + cc] * ib * invb[cc] - mask;   // logits_bb
            }
            va[jj] = xa; vb[jj] = xb;
            ma = fmaxf(ma, xa); mb = fmaxf(mb, xb);
        }
        ma = wredmax(ma); mb = wredmax(mb);
        float sea = 0.0f, seb = 0.0f;
        #pragma unroll
        for (int jj = 0; jj < 16; ++jj) {
            sea += __expf(va[jj] - ma);
            seb += __expf(vb[jj] - mb);
        }
        sea = wredsum(sea); seb = wredsum(seb);
        const float lab = GAB[i * KDIM + i] * ia * ib;   // shared diagonal label logit
        accA += (ma + __logf(sea)) - lab;
        accB += (mb + __logf(seb)) - lab;
    }

    if (lane == 0) red[w] = accA + accB;
    __syncthreads();
    if (t == 0) {
        float s = 0.0f;
        #pragma unroll
        for (int i = 0; i < 8; ++i) s += red[i];
        out[0] = s / (float)KDIM + ne_loss;
    }
}

// ---------------------------------------------------------------------------
extern "C" void kernel_launch(void* const* d_in, const int* in_sizes, int n_in,
                              void* d_out, int out_size, void* d_ws, size_t ws_size,
                              hipStream_t stream) {
    (void)in_sizes; (void)n_in; (void)out_size; (void)ws_size;
    const float* q  = (const float*)d_in[0];
    const float* qa = (const float*)d_in[1];
    float* out = (float*)d_out;

    char* ws = (char*)d_ws;
    const size_t probBytes = (size_t)KDIM * NROWS * sizeof(unsigned short); // 32 MB
    unsigned short* qsT  = (unsigned short*)ws;
    unsigned short* qasT = (unsigned short*)(ws + probBytes);
    float* GAA  = (float*)(ws + 2 * probBytes);
    float* GAB  = GAA + KDIM * KDIM;
    float* GBB  = GAB + KDIM * KDIM;
    float* cs_a  = GBB + KDIM * KDIM;
    float* csq_a = cs_a  + KDIM;
    float* cs_b  = csq_a + KDIM;
    float* csq_b = cs_b  + KDIM;

    const int nZero = 3 * KDIM * KDIM + 4 * KDIM;
    ccl_zero_kernel<<<128, 256, 0, stream>>>(GAA, nZero);

    ccl_softmax_T_kernel<<<NROWS / 32, 256, 0, stream>>>(q,  qsT,  cs_a, csq_a);
    ccl_softmax_T_kernel<<<NROWS / 32, 256, 0, stream>>>(qa, qasT, cs_b, csq_b);

    dim3 gg(NROWS / CHUNK, 8);
    ccl_gram_kernel<<<gg, 256, 0, stream>>>(qsT,  qsT,  GAA);
    ccl_gram_kernel<<<gg, 256, 0, stream>>>(qsT,  qasT, GAB);
    ccl_gram_kernel<<<gg, 256, 0, stream>>>(qasT, qasT, GBB);

    ccl_loss_kernel<<<1, 256, 0, stream>>>(GAA, GAB, GBB,
                                           cs_a, csq_a, cs_b, csq_b, out);
}
